// SPOTModel_74423193305284
// MI455X (gfx1250) — matrix operand order repaired and checked by reading, not compiled, not run on hardware
//
#include <hip/hip_runtime.h>
#include <hip/hip_bf16.h>
#include <math.h>

// ---------------- problem constants ----------------
constexpr int Bc   = 128;
constexpr int Sc   = 512;
constexpr int Tc   = Sc + 1;        // 513
constexpr int Dc   = 128;
constexpr int Hc   = 4;
constexpr int HDc  = 32;
constexpr int Lc   = 4;
constexpr int DZc  = 128;
constexpr int MAXPc= 32;
constexpr int TPc  = 544;           // keys padded to 17 blocks of 32
constexpr int Mrows= Bc * Tc;       // 65664, divisible by 32
constexpr int Mb   = Mrows / 32;    // 2052 M-blocks (32 rows each)
constexpr int OUTC = (MAXPc + 2) + Dc + DZc + DZc + 1; // 419

typedef __attribute__((ext_vector_type(16))) _Float16 v16h;
typedef __attribute__((ext_vector_type(8)))  float    v8f;
typedef _Float16 h16;

// ---------------- helpers ----------------
__device__ __forceinline__ float dev_softplus(float x) {
    return (x > 20.0f) ? x : log1pf(expf(x));
}
__device__ __forceinline__ float dev_sigmoid(float x) {
    return 1.0f / (1.0f + expf(-x));
}

// A fragment (16x32 f16, row-major source, leading dim ld), contiguous K layout:
// lanes 0-15: row=l,   K 0..15
// lanes 16-31: row=l-16, K 16..31
__device__ __forceinline__ v16h load_a_frag(const h16* __restrict__ p, int ld, int lane) {
    int row   = lane & 15;
    int chunk = (lane < 16) ? 0 : 16;
    const h16* r = p + (size_t)row * ld + chunk;
    v16h a;
#pragma unroll
    for (int e = 0; e < 8; ++e) { a[e] = r[e]; a[e + 8] = r[e + 8]; }
    return a;
}

// B fragment for B = W^T with W (N x K) row-major: B[k][n] = W[n][k].
// contiguous K layout (as A): col n = lane&15; lanes 0-15 hold K=k0..k0+15, lanes 16-31 K=k0+16..k0+31
// -> 16 contiguous halfs per lane.
__device__ __forceinline__ v16h load_bT_frag(const h16* __restrict__ p, int ld, int lane) {
    const h16* r = p + (size_t)(lane & 15) * ld + ((lane >> 4) << 4);
    v16h b;
#pragma unroll
    for (int e = 0; e < 16; ++e) b[e] = r[e];
    return b;
}

// B fragment for row-major B (K x N) (attention V). Strided gather along K,
// rows clamped to rmax (P==0 on clamped rows so result unchanged).
__device__ __forceinline__ v16h load_b_rowmajor_frag(const h16* __restrict__ p, int ld,
                                                     int lane, int rbase, int rmax) {
    int n  = lane & 15;
    int k0 = rbase + ((lane >> 4) << 4);
    v16h b;
#pragma unroll
    for (int e = 0; e < 16; ++e) {
        int r = k0 + e; r = (r > rmax) ? rmax : r;
        b[e] = p[(size_t)r * ld + n];
    }
    return b;
}

// ---------------- f32 -> f16 convert ----------------
__global__ void convert_f16_kernel(const float* __restrict__ src, h16* __restrict__ dst, int n) {
    int i = blockIdx.x * 256 + threadIdx.x;
    if (i < n) dst[i] = (h16)src[i];
}

// ---------------- embedding + agg token ----------------
__global__ void embed_kernel(const float* __restrict__ d_t, const float* __restrict__ d_l,
                             const int* __restrict__ d_emb, const float* __restrict__ poi,
                             const float* __restrict__ W_time, const float* __restrict__ W_space,
                             const float* __restrict__ agg,
                             float* __restrict__ x, h16* __restrict__ xh) {
    int idx = blockIdx.x * 256 + threadIdx.x;        // over B*T*D
    if (idx >= Bc * Tc * Dc) return;
    int d  = idx & (Dc - 1);
    int bt = idx >> 7;
    int t  = bt % Tc;
    int b  = bt / Tc;
    float v;
    if (t < Sc) {
        int bs = b * Sc + t;
        int p  = d_emb[bs];
        v = d_t[bs] * W_time[d]
          + d_l[2 * bs] * W_space[2 * d] + d_l[2 * bs + 1] * W_space[2 * d + 1]
          + poi[(size_t)p * Dc + d];
    } else {
        v = agg[d];
    }
    x[idx]  = v;
    xh[idx] = (h16)v;
}

// ---------------- WMMA GEMM: Y = X(MxK,f16) * W(NxK,f16)^T + bias ----------------
// block = 128 threads (4 waves). Block tile = 32 rows x 128 cols.
// Each wave: 32x32 output (2 A-frags x 2 B-frags -> 4 WMMAs per 32-wide K step).
// ACT: 0=none 1=relu 2=softplus
template<int ACT>
__global__ void gemm_kernel(const h16* __restrict__ X, const h16* __restrict__ W,
                            const float* __restrict__ bias, int K,
                            float* __restrict__ Yf, int ldyf,
                            h16* __restrict__ Yh, int ldyh) {
    int lane = threadIdx.x & 31;
    int wv   = threadIdx.x >> 5;
    int mb   = blockIdx.x;                       // 32-row block
    int ntp  = blockIdx.y * 8 + wv * 2;          // first of this wave's 2 N-tiles

    const h16* X0 = X + (size_t)mb * 32 * K;
    const h16* X1 = X0 + (size_t)16 * K;
    const h16* W0 = W + (size_t)ntp * 16 * K;
    const h16* W1 = W0 + (size_t)16 * K;

    v8f acc00 = {}, acc01 = {}, acc10 = {}, acc11 = {};
    for (int k0 = 0; k0 < K; k0 += 32) {
        v16h a0 = load_a_frag(X0 + k0, K, lane);
        v16h a1 = load_a_frag(X1 + k0, K, lane);
        v16h b0 = load_bT_frag(W0 + k0, K, lane);
        v16h b1 = load_bT_frag(W1 + k0, K, lane);
        acc00 = __builtin_amdgcn_wmma_f32_16x16x32_f16(false, a0, false, b0, (short)0, acc00, false, false);
        acc01 = __builtin_amdgcn_wmma_f32_16x16x32_f16(false, a0, false, b1, (short)0, acc01, false, false);
        acc10 = __builtin_amdgcn_wmma_f32_16x16x32_f16(false, a1, false, b0, (short)0, acc10, false, false);
        acc11 = __builtin_amdgcn_wmma_f32_16x16x32_f16(false, a1, false, b1, (short)0, acc11, false, false);
    }

    int n0 = ntp * 16 + (lane & 15);
    int n1 = n0 + 16;
    float bv0 = bias ? bias[n0] : 0.0f;
    float bv1 = bias ? bias[n1] : 0.0f;
#pragma unroll
    for (int r = 0; r < 8; ++r) {
        int m0 = mb * 32 + r + ((lane >= 16) ? 8 : 0);
        int m1 = m0 + 16;
        float v00 = acc00[r] + bv0, v01 = acc01[r] + bv1;
        float v10 = acc10[r] + bv0, v11 = acc11[r] + bv1;
        if (ACT == 1) {
            v00 = fmaxf(v00, 0.f); v01 = fmaxf(v01, 0.f);
            v10 = fmaxf(v10, 0.f); v11 = fmaxf(v11, 0.f);
        }
        if (ACT == 2) {
            v00 = dev_softplus(v00); v01 = dev_softplus(v01);
            v10 = dev_softplus(v10); v11 = dev_softplus(v11);
        }
        if (Yf) {
            Yf[(size_t)m0 * ldyf + n0] = v00; Yf[(size_t)m0 * ldyf + n1] = v01;
            Yf[(size_t)m1 * ldyf + n0] = v10; Yf[(size_t)m1 * ldyf + n1] = v11;
        }
        if (Yh) {
            Yh[(size_t)m0 * ldyh + n0] = (h16)v00; Yh[(size_t)m0 * ldyh + n1] = (h16)v01;
            Yh[(size_t)m1 * ldyh + n0] = (h16)v10; Yh[(size_t)m1 * ldyh + n1] = (h16)v11;
        }
    }
}

// ---------------- fused attention: one wave per (16-query tile, b*h) ----------------
// qkv: (B*T, 3D) f16. Oh: (B*T, D) f16. Dynamic LDS: 16x544 f32 scores + 16x544 f16 P.
__global__ void attn_kernel(const h16* __restrict__ qkv, const unsigned char* __restrict__ pad,
                            h16* __restrict__ Oh) {
    extern __shared__ char smem[];
    float* sc = (float*)smem;                         // [16][TPc]
    h16*   sp = (h16*)(smem + 16 * TPc * sizeof(float));

    const int ldq = 3 * Dc;                           // 384
    int qt   = blockIdx.x;                            // 0..32
    int bh   = blockIdx.y;                            // 0..511
    int b    = bh >> 2;
    int h    = bh & 3;
    int lane = threadIdx.x;

    const h16* base = qkv + (size_t)b * Tc * ldq;
    const h16* Q = base + h * HDc;
    const h16* K = base + Dc + h * HDc;
    const h16* V = base + 2 * Dc + h * HDc;

    // A fragment for this 16-query tile (rows clamped in-bounds)
    int q     = qt * 16 + (lane & 15);
    int qc    = (q < Tc) ? q : (Tc - 1);
    int chunk = (lane < 16) ? 0 : 16;
    const h16* qp = Q + (size_t)qc * ldq + chunk;
    v16h aq;
#pragma unroll
    for (int e = 0; e < 8; ++e) { aq[e] = qp[e]; aq[e + 8] = qp[e + 8]; }

    const float scale = 0.17677669529663687f;         // 1/sqrt(32)

    // scores: one WMMA per 16-key tile; mask folded into a branch-free additive term
    for (int kt = 0; kt < TPc / 16; ++kt) {
        int key  = kt * 16 + (lane & 15);
        int keyc = (key < Tc) ? key : (Tc - 1);
        // hoisted mask: one byte load + one select per key tile (uniform over r)
        float madd = ((key >= Tc) | (int)pad[b * Tc + keyc]) ? -1e30f : 0.0f;
        const h16* kp = K + (size_t)keyc * ldq + ((lane >> 4) << 4);
        v16h bk;
#pragma unroll
        for (int e = 0; e < 16; ++e) bk[e] = kp[e];
        v8f c = {};
        c = __builtin_amdgcn_wmma_f32_16x16x32_f16(false, aq, false, bk,
                                                   (short)0, c, false, false);
#pragma unroll
        for (int r = 0; r < 8; ++r) {
            int qr = r + ((lane >= 16) ? 8 : 0);
            sc[qr * TPc + key] = c[r] * scale + madd;
        }
    }
    __syncthreads();

    // softmax: lane pair (l, l+16) shares row l&15, each scans half the keys,
    // combined via wave32 cross-lane xor-shuffle.
    {
        int row = lane & 15;
        int j0  = (lane >> 4) * (TPc / 2);
        float* srow = sc + row * TPc;
        float m = -1e30f;
        for (int j = j0; j < j0 + TPc / 2; ++j) m = fmaxf(m, srow[j]);
        m = fmaxf(m, __shfl_xor(m, 16, 32));
        float sum = 0.0f;
        for (int j = j0; j < j0 + TPc / 2; ++j) {
            float e = expf(srow[j] - m); srow[j] = e; sum += e;
        }
        sum += __shfl_xor(sum, 16, 32);
        float inv = 1.0f / sum;
        h16* prow = sp + row * TPc;
        for (int j = j0; j < j0 + TPc / 2; ++j) prow[j] = (h16)(srow[j] * inv);
    }
    __syncthreads();

    // O = P @ V : 2 WMMAs per 32-key block (head_dim = 32)
    v8f o0 = {}, o1 = {};
    for (int kb = 0; kb < TPc / 32; ++kb) {
        v16h ap  = load_a_frag(sp + kb * 32, TPc, lane);
        v16h bv0 = load_b_rowmajor_frag(V,      ldq, lane, kb * 32, Tc - 1);
        v16h bv1 = load_b_rowmajor_frag(V + 16, ldq, lane, kb * 32, Tc - 1);
        o0 = __builtin_amdgcn_wmma_f32_16x16x32_f16(false, ap, false, bv0,
                                                    (short)0, o0, false, false);
        o1 = __builtin_amdgcn_wmma_f32_16x16x32_f16(false, ap, false, bv1,
                                                    (short)0, o1, false, false);
    }

#pragma unroll
    for (int r = 0; r < 8; ++r) {
        int qr = qt * 16 + r + ((lane >= 16) ? 8 : 0);
        if (qr < Tc) {
            size_t o = ((size_t)b * Tc + qr) * Dc + h * HDc + (lane & 15);
            Oh[o]      = (h16)o0[r];
            Oh[o + 16] = (h16)o1[r];
        }
    }
}

// ---------------- residual + layernorm (in place on x), writes f32 + f16 ----------------
__global__ void add_ln_kernel(float* __restrict__ x, const float* __restrict__ y,
                              const float* __restrict__ g, const float* __restrict__ beta,
                              h16* __restrict__ xh) {
    __shared__ float red[Dc];
    __shared__ float s_mean, s_rstd;
    int row = blockIdx.x, d = threadIdx.x;
    size_t idx = (size_t)row * Dc + d;
    float v = x[idx] + y[idx];
    red[d] = v; __syncthreads();
    for (int s = Dc / 2; s > 0; s >>= 1) { if (d < s) red[d] += red[d + s]; __syncthreads(); }
    if (d == 0) s_mean = red[0] * (1.0f / Dc);
    __syncthreads();
    float c = v - s_mean;
    red[d] = c * c; __syncthreads();
    for (int s = Dc / 2; s > 0; s >>= 1) { if (d < s) red[d] += red[d + s]; __syncthreads(); }
    if (d == 0) s_rstd = rsqrtf(red[0] * (1.0f / Dc) + 1e-5f);
    __syncthreads();
    float o = c * s_rstd * g[d] + beta[d];
    x[idx]  = o;
    xh[idx] = (h16)o;
}

// ---------------- extract ctx (row T-1) -> d_out ctx region, GRU h0 ----------------
__global__ void extract_ctx_kernel(const float* __restrict__ x, float* __restrict__ out,
                                   float* __restrict__ h, h16* __restrict__ hh) {
    int idx = blockIdx.x * 256 + threadIdx.x;        // over B*D
    if (idx >= Bc * Dc) return;
    int b = idx >> 7, d = idx & (Dc - 1);
    float v = x[((size_t)b * Tc + (Tc - 1)) * Dc + d];
    out[(size_t)b * OUTC + (MAXPc + 2) + d] = v;     // ctx cols [34,162)
    h[idx]  = v;
    hh[idx] = (h16)v;
}

// ---------------- GRU pointwise step + delta dot ----------------
__global__ void gru_step_kernel(const float* __restrict__ gh, const float* __restrict__ bih,
                                const float* __restrict__ Wfc, const float* __restrict__ bfc,
                                float* __restrict__ h, h16* __restrict__ hh,
                                float* __restrict__ delta) {
    __shared__ float red[Dc];
    int b = blockIdx.x, d = threadIdx.x;
    const float* g = gh + (size_t)b * (3 * Dc);
    float hv = h[(size_t)b * Dc + d];
    float r  = dev_sigmoid(bih[d] + g[d]);
    float z  = dev_sigmoid(bih[Dc + d] + g[Dc + d]);
    float n  = tanhf(bih[2 * Dc + d] + r * g[2 * Dc + d]);
    float hn = (1.0f - z) * n + z * hv;
    h[(size_t)b * Dc + d]  = hn;
    hh[(size_t)b * Dc + d] = (h16)hn;
    red[d] = hn * Wfc[d]; __syncthreads();
    for (int s = Dc / 2; s > 0; s >>= 1) { if (d < s) red[d] += red[d + s]; __syncthreads(); }
    if (d == 0) delta[b] = dev_softplus(red[0] + bfc[0]);
}

// ---------------- cumsum / times / KL assembly ----------------
__global__ void finalize_kernel(const float* __restrict__ deltas,   // (MAXP, B)
                                const int* __restrict__ num_pred,
                                float* __restrict__ out) {
    __shared__ float cum[MAXPc];
    __shared__ float red[64];
    int b = blockIdx.x, t = threadIdx.x;              // 64 threads
    if (t < MAXPc) cum[t] = deltas[(size_t)t * Bc + b];
    __syncthreads();
    if (t == 0) { float a = 0.f; for (int j = 0; j < MAXPc; ++j) { a += cum[j]; cum[j] = a; } }
    __syncthreads();
    int np = num_pred[b];
    float last = cum[np - 1];
    float* row = out + (size_t)b * OUTC;
    if (t < MAXPc + 2) {
        float v = 0.0f;
        if (t >= 1 && t <= MAXPc && (t - 1) < np) v = cum[t - 1] / (last + 1e-6f);
        if (t == np + 1) v += 1.0f;
        row[t] = v;
    }
    const float* gamma = row + (MAXPc + 2) + Dc;
    const float* tau   = gamma + DZc;
    float p = 0.0f;
    for (int d = t; d < DZc; d += 64) {
        float tv = tau[d], gv = gamma[d];
        p += tv * tv + gv * gv - 2.0f * logf(tv);
    }
    red[t] = p; __syncthreads();
    for (int s = 32; s > 0; s >>= 1) { if (t < s) red[t] += red[t + s]; __syncthreads(); }
    if (t == 0) row[OUTC - 1] = 0.5f * (red[0] - (float)DZc);
}

// ---------------- launch ----------------
extern "C" void kernel_launch(void* const* d_in, const int* in_sizes, int n_in,
                              void* d_out, int out_size, void* d_ws, size_t ws_size,
                              hipStream_t stream) {
    (void)in_sizes; (void)n_in; (void)out_size; (void)ws_size;

    const float* d_t     = (const float*)d_in[0];
    const float* d_l     = (const float*)d_in[1];
    const int*   d_emb   = (const int*)d_in[2];
    const unsigned char* d_pad = (const unsigned char*)d_in[3];
    const int*   num_pred= (const int*)d_in[4];
    const float* poi     = (const float*)d_in[5];
    const float* W_time  = (const float*)d_in[6];
    const float* W_space = (const float*)d_in[7];
    const float* agg     = (const float*)d_in[8];
    const float* Wqkv    = (const float*)d_in[9];
    const float* bqkv    = (const float*)d_in[10];
    const float* Wo      = (const float*)d_in[11];
    const float* bo      = (const float*)d_in[12];
    const float* ln1_g   = (const float*)d_in[13];
    const float* ln1_b   = (const float*)d_in[14];
    const float* W1      = (const float*)d_in[15];
    const float* b1      = (const float*)d_in[16];
    const float* W2      = (const float*)d_in[17];
    const float* b2      = (const float*)d_in[18];
    const float* ln2_g   = (const float*)d_in[19];
    const float* ln2_b   = (const float*)d_in[20];
    const float* Wg      = (const float*)d_in[21];
    const float* bg      = (const float*)d_in[22];
    const float* Wt      = (const float*)d_in[23];
    const float* bt      = (const float*)d_in[24];
    const float* gru_Whh = (const float*)d_in[26];
    const float* gru_bih = (const float*)d_in[27];
    const float* gru_bhh = (const float*)d_in[28];
    const float* Wfc     = (const float*)d_in[29];
    const float* bfc     = (const float*)d_in[30];
    float* out = (float*)d_out;

    // workspace carve-up
    char* wp = (char*)d_ws;
    auto carve = [&](size_t bytes) -> void* {
        void* p = (void*)wp; wp += (bytes + 255) & ~(size_t)255; return p;
    };
    float* x     = (float*)carve((size_t)Mrows * Dc * 4);
    h16*   Xh    = (h16*)  carve((size_t)Mrows * Dc * 2);
    h16*   QKVh  = (h16*)  carve((size_t)Mrows * 3 * Dc * 2);
    h16*   Oh    = (h16*)  carve((size_t)Mrows * Dc * 2);
    float* Yt    = (float*)carve((size_t)Mrows * Dc * 4);
    h16*   H1h   = (h16*)  carve((size_t)Mrows * 2 * Dc * 2);
    h16*   Wqkvh = (h16*)  carve((size_t)Lc * 3 * Dc * Dc * 2);
    h16*   Woh   = (h16*)  carve((size_t)Lc * Dc * Dc * 2);
    h16*   W1h   = (h16*)  carve((size_t)Lc * 2 * Dc * Dc * 2);
    h16*   W2h   = (h16*)  carve((size_t)Lc * Dc * 2 * Dc * 2);
    h16*   Wgh   = (h16*)  carve((size_t)DZc * Dc * 2);
    h16*   Wth   = (h16*)  carve((size_t)DZc * Dc * 2);
    h16*   Whhh  = (h16*)  carve((size_t)3 * Dc * Dc * 2);
    float* ghb   = (float*)carve((size_t)Bc * 3 * Dc * 4);
    float* hbuf  = (float*)carve((size_t)Bc * Dc * 4);
    h16*   hh    = (h16*)  carve((size_t)Bc * Dc * 2);
    float* deltas= (float*)carve((size_t)MAXPc * Bc * 4);

    auto conv = [&](const float* s, h16* d, int n) {
        convert_f16_kernel<<<(n + 255) / 256, 256, 0, stream>>>(s, d, n);
    };
    conv(Wqkv, Wqkvh, Lc * 3 * Dc * Dc);
    conv(Wo,   Woh,   Lc * Dc * Dc);
    conv(W1,   W1h,   Lc * 2 * Dc * Dc);
    conv(W2,   W2h,   Lc * Dc * 2 * Dc);
    conv(Wg,   Wgh,   DZc * Dc);
    conv(Wt,   Wth,   DZc * Dc);
    conv(gru_Whh, Whhh, 3 * Dc * Dc);

    // embeddings -> x, Xh
    {
        int n = Bc * Tc * Dc;
        embed_kernel<<<(n + 255) / 256, 256, 0, stream>>>(d_t, d_l, d_emb, poi,
                                                          W_time, W_space, agg, x, Xh);
    }

    const size_t attn_lds = 16 * TPc * sizeof(float) + 16 * TPc * sizeof(h16);

    for (int i = 0; i < Lc; ++i) {
        const h16* Wq_i = Wqkvh + (size_t)i * 3 * Dc * Dc;
        const h16* Wo_i = Woh   + (size_t)i * Dc * Dc;
        const h16* W1_i = W1h   + (size_t)i * 2 * Dc * Dc;
        const h16* W2_i = W2h   + (size_t)i * Dc * 2 * Dc;

        // QKV = Xh @ Wqkv^T + bqkv -> f16 (M x 384); block tile 32x128
        gemm_kernel<0><<<dim3(Mb, 3), 128, 0, stream>>>(
            Xh, Wq_i, bqkv + i * 3 * Dc, Dc, nullptr, 0, QKVh, 3 * Dc);

        // fused attention -> Oh (M x 128) f16
        attn_kernel<<<dim3((Tc + 15) / 16, Bc * Hc), 32, attn_lds, stream>>>(QKVh, d_pad, Oh);

        // Y = Oh @ Wo^T + bo -> f32
        gemm_kernel<0><<<dim3(Mb, 1), 128, 0, stream>>>(
            Oh, Wo_i, bo + i * Dc, Dc, Yt, Dc, nullptr, 0);

        // x = LN(x + Y)
        add_ln_kernel<<<Mrows, Dc, 0, stream>>>(x, Yt, ln1_g + i * Dc, ln1_b + i * Dc, Xh);

        // H1 = relu(Xh @ W1^T + b1) -> f16 (M x 256)
        gemm_kernel<1><<<dim3(Mb, 2), 128, 0, stream>>>(
            Xh, W1_i, b1 + i * 2 * Dc, Dc, nullptr, 0, H1h, 2 * Dc);

        // Y = H1 @ W2^T + b2 -> f32
        gemm_kernel<0><<<dim3(Mb, 1), 128, 0, stream>>>(
            H1h, W2_i, b2 + i * Dc, 2 * Dc, Yt, Dc, nullptr, 0);

        // x = LN(x + Y)
        add_ln_kernel<<<Mrows, Dc, 0, stream>>>(x, Yt, ln2_g + i * Dc, ln2_b + i * Dc, Xh);
    }

    // ctx -> d_out[:,34:162], GRU h0
    extract_ctx_kernel<<<(Bc * Dc + 255) / 256, 256, 0, stream>>>(x, out, hbuf, hh);

    // gamma = ctx @ Wg^T + bg -> d_out cols [162,290), strided ld = 419
    gemm_kernel<0><<<dim3(Bc / 32, 1), 128, 0, stream>>>(
        hh, Wgh, bg, Dc, out + (MAXPc + 2) + Dc, OUTC, nullptr, 0);
    // tau = softplus(ctx @ Wt^T + bt) -> d_out cols [290,418)
    gemm_kernel<2><<<dim3(Bc / 32, 1), 128, 0, stream>>>(
        hh, Wth, bt, Dc, out + (MAXPc + 2) + Dc + DZc, OUTC, nullptr, 0);

    // GRU scan: 32 sequential (GEMM + pointwise) steps
    for (int step = 0; step < MAXPc; ++step) {
        gemm_kernel<0><<<dim3(Bc / 32, 3), 128, 0, stream>>>(
            hh, Whhh, gru_bhh, Dc, ghb, 3 * Dc, nullptr, 0);
        gru_step_kernel<<<Bc, Dc, 0, stream>>>(ghb, gru_bih, Wfc, bfc,
                                               hbuf, hh, deltas + (size_t)step * Bc);
    }

    // times + KL -> remaining output columns
    finalize_kernel<<<Bc, 64, 0, stream>>>(deltas, num_pred, out);
}
